// Ns_Transformer_46849503265152
// MI455X (gfx1250) — matrix operand order repaired and checked by reading, not compiled, not run on hardware
//
#include <hip/hip_runtime.h>
#include <hip/hip_bf16.h>
#include <math.h>

// Problem constants (match reference)
#define BB    256
#define SS    120
#define DDIM  1024
#define HH    8
#define HDIM  128
#define RBNN  4096
#define BHEAD (BB * HH)          // 2048
#define MROWS (BB * SS)          // 30720
#define SPAD  128                // S padded to 128 for WMMA

typedef __attribute__((ext_vector_type(16))) __bf16 v16bf;
typedef __attribute__((ext_vector_type(8)))  float  v8f;
typedef __attribute__((ext_vector_type(4))) unsigned int tdm_v4u;
typedef __attribute__((ext_vector_type(8))) int          tdm_v8i;
typedef __attribute__((ext_vector_type(4))) int          tdm_v4i;

#if defined(__has_builtin)
#if __has_builtin(__builtin_amdgcn_tensor_load_to_lds)
#define HAVE_TDM 1
#endif
#endif
#ifndef HAVE_TDM
#define HAVE_TDM 0
#endif

union FragBF { v16bf v; unsigned int u[8]; };

__device__ __forceinline__ unsigned short f32_to_bf16(float f) {
  unsigned int u = __float_as_uint(f);
  u += 0x7FFFu + ((u >> 16) & 1u);        // round-to-nearest-even
  return (unsigned short)(u >> 16);
}

// A-matrix fragment (16x32 bf16, row-major in LDS, row = M):
// dword i of lane holds K0 = half*16 + i*2 (contiguous, matches load_frag_b)
__device__ __forceinline__ v16bf load_frag_a(const unsigned int* lds, int row,
                                             int strideDw, int kBaseDw, int half) {
  FragBF f;
  const unsigned int* rp = lds + row * strideDw + kBaseDw;
#pragma unroll
  for (int i = 0; i < 8; ++i) {
    int k0 = (half << 4) + (i << 1);
    f.u[i] = rp[k0 >> 1];
  }
  return f.v;
}

// B-matrix fragment (32x16 bf16) from an N-major [n][32] LDS tile:
// lane -> N = l&15, element j -> K = half*16 + j  => 8 contiguous dwords.
__device__ __forceinline__ v16bf load_frag_b(const unsigned int* lds, int nrow,
                                             int strideDw, int kBaseDw, int half) {
  FragBF f;
  const unsigned int* rp = lds + nrow * strideDw + kBaseDw + (half << 3);
#pragma unroll
  for (int i = 0; i < 8; ++i) f.u[i] = rp[i];
  return f.v;
}

#if HAVE_TDM
// TDM 2D tile load: [tile1 rows x tile0 elems] of 2-byte data, row stride
// stride0 elements, into contiguous LDS at ldsAddr. D# per ISA 08 §8.3/8.4.
__device__ __forceinline__ void tdm_load_2d(unsigned int ldsAddr, const void* gptr,
                                            unsigned int dim0, unsigned int dim1,
                                            unsigned int tile0, unsigned int tile1,
                                            unsigned long long stride0) {
  unsigned long long ga = (unsigned long long)(size_t)gptr;
  tdm_v4u g0;
  g0.x = 1u;                                                   // count=1, user mode
  g0.y = ldsAddr;                                              // lds_addr (bytes)
  g0.z = (unsigned int)ga;                                     // global_addr[31:0]
  g0.w = (unsigned int)((ga >> 32) & 0x01FFFFFFu) | (2u << 30);// addr[56:32] | type=2
  tdm_v8i g1;
  g1[0] = (int)(1u << 16);                                     // data_size=1 (2 bytes)
  g1[1] = (int)((dim0 & 0xFFFFu) << 16);                       // tensor_dim0[15:0]
  g1[2] = (int)(((dim0 >> 16) & 0xFFFFu) | ((dim1 & 0xFFFFu) << 16));
  g1[3] = (int)(((dim1 >> 16) & 0xFFFFu) | ((tile0 & 0xFFFFu) << 16));
  g1[4] = (int)(tile1 & 0xFFFFu);                              // tile_dim1, tile_dim2=0
  g1[5] = (int)(unsigned int)(stride0 & 0xFFFFFFFFull);        // dim0_stride[31:0]
  g1[6] = (int)(unsigned int)((stride0 >> 32) & 0xFFFFull);    // dim0_stride[47:32]
  g1[7] = 0;
  tdm_v4i z4 = (tdm_v4i)0;
#if __clang_major__ >= 23
  tdm_v8i z8 = (tdm_v8i)0;
  __builtin_amdgcn_tensor_load_to_lds(g0, g1, z4, z4, z8, 0);
#else
  __builtin_amdgcn_tensor_load_to_lds(g0, g1, z4, z4, 0);
#endif
}
#endif

// ---------------------------------------------------------------------------
// fp32 -> bf16 (grid-stride)
// ---------------------------------------------------------------------------
__global__ void cvt_f32_bf16(const float* __restrict__ in,
                             unsigned short* __restrict__ out, size_t n) {
  size_t i = (size_t)blockIdx.x * blockDim.x + threadIdx.x;
  size_t stride = (size_t)gridDim.x * blockDim.x;
  for (; i < n; i += stride) out[i] = f32_to_bf16(in[i]);
}

// fp32 [K][N] -> bf16 [N][K] (weights pre-transposed so GEMM B tiles are
// row-major in K, TDM/b128-friendly, no in-kernel transpose)
__global__ void cvt_transpose_bf16(const float* __restrict__ in,
                                   unsigned short* __restrict__ out, int K, int N) {
  size_t n = (size_t)K * N;
  size_t i = (size_t)blockIdx.x * blockDim.x + threadIdx.x;
  size_t stride = (size_t)gridDim.x * blockDim.x;
  for (; i < n; i += stride) {
    size_t k = i / (size_t)N;
    size_t c = i - k * (size_t)N;
    out[c * (size_t)K + k] = f32_to_bf16(in[i]);
  }
}

// zero the pad regions of padded q/k [head][128][128] (rows 120..127) and
// vT [head][128 hd][128 key] (keys 120..127)
__global__ void zero_pads(unsigned short* __restrict__ qb,
                          unsigned short* __restrict__ kb,
                          unsigned short* __restrict__ vt) {
  size_t n = (size_t)BHEAD * 8 * SPAD;
  size_t i = (size_t)blockIdx.x * blockDim.x + threadIdx.x;
  size_t stride = (size_t)gridDim.x * blockDim.x;
  for (; i < n; i += stride) {
    size_t head = i >> 10;              // 8*128 per head
    int r = (int)((i >> 7) & 7);
    int c = (int)(i & 127);
    qb[(head * SPAD + SS + r) * SPAD + c] = 0;
    kb[(head * SPAD + SS + r) * SPAD + c] = 0;
    vt[(head * SPAD + c) * SPAD + SS + r] = 0;
  }
}

// ---------------------------------------------------------------------------
// Tiled WMMA GEMM: C[M,N] = A[M,K] * Bt[N,K]^T (+bias, epilogue per MODE)
//   BM=BN=128, BK=32, 256 threads = 8 waves (2 over M x 4 over N),
//   wave tile 64x32 = 8 accumulators, 8 v_wmma per K-step.
//   Tiles staged by TDM (double-buffered, TENSORcnt-pipelined) when available.
// MODE 0: QKV scatter -> q/k [head][128][128] bf16 + vT [head][hd][key] bf16
// MODE 1: bias + exact GELU -> bf16 row-major
// MODE 2: bias -> fp32 row-major (final output)
// ---------------------------------------------------------------------------
template <int MODE>
__global__ __launch_bounds__(256, 1) void gemm_bf16_wmma(
    const unsigned short* __restrict__ A, const unsigned short* __restrict__ Bt,
    const float* __restrict__ bias, int M, int N, int K,
    float* __restrict__ outF, unsigned short* __restrict__ outB,
    unsigned short* __restrict__ outQ, unsigned short* __restrict__ outK,
    unsigned short* __restrict__ outV) {
  __shared__ __align__(16) unsigned int As[2][128 * 16];  // 128 rows x 32 bf16
  __shared__ __align__(16) unsigned int Bs[2][128 * 16];  // 128 n-rows x 32 bf16

  const int tid  = threadIdx.x;
  const int lane = tid & 31;
  const int half = lane >> 4;
  const int ln16 = lane & 15;
  const int wid  = tid >> 5;
  const int wm   = wid & 1;    // 2 waves over M, 64 rows each
  const int wn   = wid >> 1;   // 4 waves over N, 32 cols each

  const int rowBase = blockIdx.y * 128;
  const int colBase = blockIdx.x * 128;
  const int nk = K >> 5;

  v8f acc[4][2] = {};

#if HAVE_TDM
  unsigned int ldsA[2], ldsB[2];
  ldsA[0] = (unsigned int)(size_t)&As[0][0];
  ldsA[1] = (unsigned int)(size_t)&As[1][0];
  ldsB[0] = (unsigned int)(size_t)&Bs[0][0];
  ldsB[1] = (unsigned int)(size_t)&Bs[1][0];
  if (wid == 0) {
    tdm_load_2d(ldsA[0], A + (size_t)rowBase * K, K, 128, 32, 128, (unsigned long long)K);
    tdm_load_2d(ldsB[0], Bt + (size_t)colBase * K, K, 128, 32, 128, (unsigned long long)K);
  }
  for (int kt = 0; kt < nk; ++kt) {
    const int cur = kt & 1;
    if (wid == 0) {
      if (kt + 1 < nk) {
        const int k1 = (kt + 1) << 5;
        tdm_load_2d(ldsA[cur ^ 1], A + (size_t)rowBase * K + k1, K, 128, 32, 128,
                    (unsigned long long)K);
        tdm_load_2d(ldsB[cur ^ 1], Bt + (size_t)colBase * K + k1, K, 128, 32, 128,
                    (unsigned long long)K);
        __builtin_amdgcn_s_wait_tensorcnt(2);   // current pair landed
      } else {
        __builtin_amdgcn_s_wait_tensorcnt(0);
      }
    }
    __syncthreads();
    const unsigned int* Ab = As[cur];
    const unsigned int* Bb = Bs[cur];
#else
  const unsigned int* A32 = (const unsigned int*)A;
  const unsigned int* B32 = (const unsigned int*)Bt;
  const int KD = K >> 1;
  for (int kt = 0; kt < nk; ++kt) {
    const int k0 = kt << 5;
    // fallback staging: 2 uint4 per thread per tile
#pragma unroll
    for (int c = 0; c < 2; ++c) {
      const int chunk = tid + c * 256;
      const int row = chunk >> 2;
      const int dwc = (chunk & 3) << 2;
      ((uint4*)As[0])[chunk] =
          *(const uint4*)(A32 + (size_t)(rowBase + row) * KD + (k0 >> 1) + dwc);
      ((uint4*)Bs[0])[chunk] =
          *(const uint4*)(B32 + (size_t)(colBase + row) * KD + (k0 >> 1) + dwc);
    }
    __syncthreads();
    const unsigned int* Ab = As[0];
    const unsigned int* Bb = Bs[0];
#endif
    v16bf af[4], bfg[2];
#pragma unroll
    for (int s = 0; s < 4; ++s)
      af[s] = load_frag_a(Ab, wm * 64 + s * 16 + ln16, 16, 0, half);
#pragma unroll
    for (int j = 0; j < 2; ++j)
      bfg[j] = load_frag_b(Bb, wn * 32 + j * 16 + ln16, 16, 0, half);
#pragma unroll
    for (int s = 0; s < 4; ++s)
#pragma unroll
      for (int j = 0; j < 2; ++j)
        acc[s][j] = __builtin_amdgcn_wmma_f32_16x16x32_bf16(
            false, af[s], false, bfg[j], (short)0, acc[s][j], false, false);
    __syncthreads();
  }

  // Epilogue. D layout: lane -> N = ln16, VGPR r -> M = r + 8*half.
#pragma unroll
  for (int s = 0; s < 4; ++s) {
    const int mBase = rowBase + wm * 64 + s * 16 + 8 * half;
#pragma unroll
    for (int j = 0; j < 2; ++j) {
      const int ncol = colBase + wn * 32 + j * 16 + ln16;
      const float bv2 = bias[ncol];
#pragma unroll
      for (int r = 0; r < 8; ++r) {
        const int mrow = mBase + r;
        float val = acc[s][j][r] + bv2;
        if (MODE == 0) {
          const int which = ncol >> 10;          // 0=q 1=k 2=v
          const int c  = ncol & 1023;
          const int h  = c >> 7;
          const int hd = c & 127;
          const int b  = mrow / SS;
          const int sq = mrow - b * SS;
          const size_t head = (size_t)((b << 3) + h);
          const unsigned short bb = f32_to_bf16(val);
          if (which == 0)      outQ[(head * SPAD + sq) * SPAD + hd] = bb;
          else if (which == 1) outK[(head * SPAD + sq) * SPAD + hd] = bb;
          else                 outV[(head * SPAD + hd) * SPAD + sq] = bb; // vT
        } else if (MODE == 1) {
          const float g = 0.5f * val * (1.0f + erff(val * 0.70710678118654752f));
          outB[(size_t)mrow * N + ncol] = f32_to_bf16(g);
        } else {
          outF[(size_t)mrow * N + ncol] = val;
        }
      }
    }
  }
}

// ---------------------------------------------------------------------------
// Attention: one workgroup per head. Q,K [128][128] row-major and V^T
// [hd][key] staged into LDS with three TDM loads. Wave w owns query rows
// [16w,16w+16): scores -> softmax -> P (bf16 into own Q rows) -> P@V.
// ---------------------------------------------------------------------------
__global__ __launch_bounds__(256, 1) void attn_wmma(
    const unsigned short* __restrict__ q, const unsigned short* __restrict__ k,
    const unsigned short* __restrict__ vt, const float* __restrict__ tau,
    const float* __restrict__ delta, unsigned short* __restrict__ xout) {
  __shared__ __align__(16) unsigned int Qs[SPAD * 64];  // [row][hd]
  __shared__ __align__(16) unsigned int Ks[SPAD * 64];  // [key][hd]
  __shared__ __align__(16) unsigned int Vt[SPAD * 64];  // [hd][key]

  const int head = blockIdx.x;
  const int tid  = threadIdx.x;
  const int lane = tid & 31;
  const int half = lane >> 4;
  const int ln16 = lane & 15;
  const int wid  = tid >> 5;
  const int m0   = wid * 16;

  const size_t base = (size_t)head * SPAD * SPAD;
#if HAVE_TDM
  if (wid == 0) {
    tdm_load_2d((unsigned int)(size_t)&Qs[0], q + base, SPAD, SPAD, SPAD, SPAD, SPAD);
    tdm_load_2d((unsigned int)(size_t)&Ks[0], k + base, SPAD, SPAD, SPAD, SPAD, SPAD);
    tdm_load_2d((unsigned int)(size_t)&Vt[0], vt + base, SPAD, SPAD, SPAD, SPAD, SPAD);
    __builtin_amdgcn_s_wait_tensorcnt(0);
  }
#else
  {
    const unsigned int* q32 = (const unsigned int*)q + (base >> 1);
    const unsigned int* k32 = (const unsigned int*)k + (base >> 1);
    const unsigned int* v32 = (const unsigned int*)vt + (base >> 1);
    for (int i = tid; i < SPAD * 64; i += 256) {
      Qs[i] = q32[i]; Ks[i] = k32[i]; Vt[i] = v32[i];
    }
  }
#endif
  __syncthreads();

  const float scale2 = tau[head] * 0.08838834764831845f;  // HD^-0.5 * tau
  const float dbias  = delta[head];

  v16bf qf[4];
#pragma unroll
  for (int kc = 0; kc < 4; ++kc) qf[kc] = load_frag_a(Qs, m0 + ln16, 64, kc * 16, half);

  // scores S = (Q K^T) * scale2 + delta   (8 key-blocks x 4 k-chunks of 32)
  v8f sc[8];
#pragma unroll
  for (int j = 0; j < 8; ++j) {
    v8f acc = {};
#pragma unroll
    for (int kc = 0; kc < 4; ++kc) {
      v16bf bf = load_frag_b(Ks, j * 16 + ln16, 64, kc * 16, half);
      acc = __builtin_amdgcn_wmma_f32_16x16x32_bf16(false, qf[kc], false, bf,
                                                    (short)0, acc, false, false);
    }
    const int key = j * 16 + ln16;
    const float mask = (key < SS) ? 0.0f : -1.0e30f;
#pragma unroll
    for (int r = 0; r < 8; ++r) acc[r] = acc[r] * scale2 + dbias + mask;
    sc[j] = acc;
  }

  // softmax per row (row lives in 16 lanes of one half-wave; xor<=8 stays inside)
  float rinv[8];
#pragma unroll
  for (int r = 0; r < 8; ++r) {
    float m = -3.0e38f;
#pragma unroll
    for (int j = 0; j < 8; ++j) m = fmaxf(m, sc[j][r]);
#pragma unroll
    for (int off = 8; off >= 1; off >>= 1) m = fmaxf(m, __shfl_xor(m, off, 32));
    float s = 0.0f;
#pragma unroll
    for (int j = 0; j < 8; ++j) { float e = __expf(sc[j][r] - m); sc[j][r] = e; s += e; }
#pragma unroll
    for (int off = 8; off >= 1; off >>= 1) s += __shfl_xor(s, off, 32);
    rinv[r] = 1.0f / s;
  }

  // write normalized P (bf16) into this wave's own Q rows in LDS
  {
    unsigned short* Qsu = (unsigned short*)Qs;
#pragma unroll
    for (int j = 0; j < 8; ++j) {
      const int key = j * 16 + ln16;
#pragma unroll
      for (int r = 0; r < 8; ++r) {
        const int row = m0 + r + 8 * half;
        Qsu[row * SPAD + key] = f32_to_bf16(sc[j][r] * rinv[r]);
      }
    }
  }
  __syncthreads();

  // x = P @ V   (8 hd-blocks x 4 key-chunks), scatter to [B,S,D] bf16
  v16bf pf[4];
#pragma unroll
  for (int kc = 0; kc < 4; ++kc) pf[kc] = load_frag_a(Qs, m0 + ln16, 64, kc * 16, half);

  const int bidx = head >> 3;
  const int hidx = head & 7;
#pragma unroll
  for (int j = 0; j < 8; ++j) {
    v8f acc = {};
#pragma unroll
    for (int kc = 0; kc < 4; ++kc) {
      v16bf bf = load_frag_b(Vt, j * 16 + ln16, 64, kc * 16, half);
      acc = __builtin_amdgcn_wmma_f32_16x16x32_bf16(false, pf[kc], false, bf,
                                                    (short)0, acc, false, false);
    }
    const int col = (hidx << 7) + j * 16 + ln16;
#pragma unroll
    for (int r = 0; r < 8; ++r) {
      const int srow = m0 + r + 8 * half;
      if (srow < SS)
        xout[((size_t)bidx * SS + srow) * DDIM + col] = f32_to_bf16(acc[r]);
    }
  }
}

// ---------------------------------------------------------------------------
extern "C" void kernel_launch(void* const* d_in, const int* in_sizes, int n_in,
                              void* d_out, int out_size, void* d_ws, size_t ws_size,
                              hipStream_t stream) {
  (void)in_sizes; (void)n_in; (void)out_size; (void)ws_size;
  const float* data  = (const float*)d_in[0];
  const float* tau   = (const float*)d_in[1];
  const float* delta = (const float*)d_in[2];
  const float* Wqkv  = (const float*)d_in[3];
  const float* bqkv  = (const float*)d_in[4];
  const float* W1    = (const float*)d_in[5];
  const float* b1    = (const float*)d_in[6];
  const float* W2    = (const float*)d_in[7];
  const float* b2    = (const float*)d_in[8];
  float* out = (float*)d_out;

  char* ws = (char*)d_ws;
  size_t off = 0;
  auto carve = [&](size_t elems) {
    unsigned short* p = (unsigned short*)(ws + off);
    off = (off + elems * sizeof(unsigned short) + 255) & ~(size_t)255;
    return p;
  };
  unsigned short* Xbf   = carve((size_t)MROWS * DDIM);
  unsigned short* WqkvT = carve((size_t)3 * DDIM * DDIM);   // [3072][1024]
  unsigned short* W1T   = carve((size_t)RBNN * DDIM);       // [4096][1024]
  unsigned short* W2T   = carve((size_t)DDIM * RBNN);       // [1024][4096]
  unsigned short* qb    = carve((size_t)BHEAD * SPAD * SPAD);
  unsigned short* kb    = carve((size_t)BHEAD * SPAD * SPAD);
  unsigned short* vtb   = carve((size_t)BHEAD * SPAD * SPAD);
  unsigned short* xab   = carve((size_t)MROWS * DDIM);
  unsigned short* Hb    = carve((size_t)MROWS * RBNN);

  // staging: bf16 conversion, weight transpose, pad zeroing
  cvt_f32_bf16<<<2048, 256, 0, stream>>>(data, Xbf, (size_t)MROWS * DDIM);
  cvt_transpose_bf16<<<1024, 256, 0, stream>>>(Wqkv, WqkvT, DDIM, 3 * DDIM);
  cvt_transpose_bf16<<<1024, 256, 0, stream>>>(W1, W1T, DDIM, RBNN);
  cvt_transpose_bf16<<<1024, 256, 0, stream>>>(W2, W2T, RBNN, DDIM);
  zero_pads<<<1024, 256, 0, stream>>>(qb, kb, vtb);

  // QKV projection + head-major (padded) scatter
  gemm_bf16_wmma<0><<<dim3(3 * DDIM / 128, MROWS / 128), 256, 0, stream>>>(
      Xbf, WqkvT, bqkv, MROWS, 3 * DDIM, DDIM, nullptr, nullptr, qb, kb, vtb);

  // attention (one workgroup per head)
  attn_wmma<<<BHEAD, 256, 0, stream>>>(qb, kb, vtb, tau, delta, xab);

  // MLP: GELU(x @ W1 + b1) @ W2 + b2
  gemm_bf16_wmma<1><<<dim3(RBNN / 128, MROWS / 128), 256, 0, stream>>>(
      xab, W1T, b1, MROWS, RBNN, DDIM, nullptr, Hb, nullptr, nullptr, nullptr);
  gemm_bf16_wmma<2><<<dim3(DDIM / 128, MROWS / 128), 256, 0, stream>>>(
      Hb, W2T, b2, MROWS, DDIM, RBNN, out, nullptr, nullptr, nullptr, nullptr);
}
